// MPGNN_18073222382245
// MI455X (gfx1250) — compile-verified, matrix operand order checked by reading, not run on hardware
//
#include <hip/hip_runtime.h>
#include <hip/hip_bf16.h>

// ---------------------------------------------------------------------------
// MPGNN forward for MI455X (gfx1250, wave32, WMMA).
// All dense layers run as bf16 WMMA 16x16x32 GEMMs with fp32 accumulation.
// Segment widths are forced to multiples of 32 so the concat-segment select
// in the A-tile loader is wave-uniform (scalar branches, EXEC stays all-1s).
// ---------------------------------------------------------------------------

typedef __bf16 bf16;
typedef bf16  v16bf __attribute__((ext_vector_type(16)));
typedef bf16  v8bf  __attribute__((ext_vector_type(8)));
typedef float v8f   __attribute__((ext_vector_type(8)));

#define NN 50000      // nodes  (50000 = 3125*16)
#define EE 800000     // edges  (800000 = 50000*16)
#define GG 64         // graphs (64 = 4*16)

// ---------------------------------------------------------------------------
// Generic fused GEMM:  out[M x Nout] = preop(concat(seg0,seg1,seg2)) @ W^T + b
//  - A: gathered/concatenated bf16, widths multiples of 32, sum == Kp
//  - W: fp32 in global, staged bf16 into LDS (zero-padded past Kw)
//  - optional BN(scale,shift)+ReLU pre-op on A (scale/shift staged in LDS)
//  - optional scatter-add epilogue (segment_sum fused into node1's GEMM2)
// ---------------------------------------------------------------------------
struct GemmArgs {
  const bf16* a0; const int* g0; int w0; int ld0;
  const bf16* a1; const int* g1; int w1; int ld1;
  const bf16* a2; const int* g2; int w2; int ld2;
  int Kp;              // total K, multiple of 32 (== w0+w1+w2)
  const float* W;      // [Nout x Kw] row-major fp32 (fan_out x fan_in)
  int Kw;              // true fan_in (may be < Kp; weights zero-padded in LDS)
  const float* bias;   // [Nout]
  const float* bn_scale; // [Kp] or null
  const float* bn_shift; // [Kp] or null
  bf16*  out_bf;       // [M x ldo] or null
  float* out_f32;      // [M x ldo] or null
  int ldo;
  const int* scat_idx; // scatter row index, or null
  float* scat_dst;     // scatter destination (atomicAdd f32), or null
  int Mtiles;
  int Nout;
};

// Load one 32x16 bf16 B tile (= W^T tile) from LDS. B lane layout mirrors A:
// lane&15 selects output column n; lane>>4 selects which K-halves
// (group 0: K {0..7,16..23}, group 1: K {8..15,24..31}); B(k,n) = W[n][k].
__device__ __forceinline__ v16bf load_b(const bf16* Wl, int Kp, int nt, int kt, int lane) {
  const int n  = nt * 16 + (lane & 15);
  const int hs = lane >> 4;
  const bf16* p = Wl + (size_t)n * Kp + kt * 32 + hs * 8;
  const v8bf c0 = *(const v8bf*)(p);        // ds_load_b128
  const v8bf c1 = *(const v8bf*)(p + 16);   // ds_load_b128
  v16bf out;
#pragma unroll
  for (int i = 0; i < 8; ++i) { out[i] = c0[i]; out[8 + i] = c1[i]; }
  return out;
}

template <int NT>  // NT = Nout/16 column tiles (8 or 4)
__global__ __launch_bounds__(256) void wmma_gemm(GemmArgs g) {
  extern __shared__ char smem[];
  bf16*  Wl = (bf16*)smem;
  float* Bl = (float*)(smem + (size_t)g.Nout * g.Kp * sizeof(bf16));
  float* Sl = Bl + g.Nout;        // BN scale [Kp]
  float* Hl = Sl + g.Kp;          // BN shift [Kp]

  // Cooperative fp32 -> bf16 weight stage into LDS (zero-padded to Kp).
  const int tidb  = threadIdx.x;
  const int total = g.Nout * g.Kp;
  for (int idx = tidb; idx < total; idx += blockDim.x) {
    const int n = idx / g.Kp, k = idx - n * g.Kp;
    Wl[idx] = (k < g.Kw) ? (bf16)g.W[(size_t)n * g.Kw + k] : (bf16)0.0f;
  }
  for (int idx = tidb; idx < g.Nout; idx += blockDim.x) Bl[idx] = g.bias[idx];
  if (g.bn_scale) {
    for (int idx = tidb; idx < g.Kp; idx += blockDim.x) {
      Sl[idx] = g.bn_scale[idx];
      Hl[idx] = g.bn_shift[idx];
    }
  }
  __syncthreads();

  const int wave = tidb >> 5;
  const int lane = tidb & 31;
  const int tile = blockIdx.x * (blockDim.x >> 5) + wave;
  if (tile >= g.Mtiles) return;   // whole-wave uniform; EXEC all-ones for WMMA

  // Hoist per-lane gather rows + base pointers (loop-invariant).
  const int r  = lane & 15;
  const int hs = lane >> 4;
  const int mbase = tile * 16;
  const int m  = mbase + r;
  const bf16* p0 = g.a0 + (size_t)(g.g0 ? g.g0[m] : m) * g.ld0;
  const bf16* p1 = g.a1 ? g.a1 + (size_t)(g.g1 ? g.g1[m] : m) * g.ld1 : nullptr;
  const bf16* p2 = g.a2 ? g.a2 + (size_t)(g.g2 ? g.g2[m] : m) * g.ld2 : nullptr;
  const int b01 = g.w0, b12 = g.w0 + g.w1;

  v8f acc[NT];
#pragma unroll
  for (int i = 0; i < NT; ++i)
#pragma unroll
    for (int j = 0; j < 8; ++j) acc[i][j] = 0.0f;

  const int KT = g.Kp >> 5;
  for (int kt = 0; kt < KT; ++kt) {
    const int kb = kt * 32;            // uniform: tile never straddles a segment
    const bf16* src; int off;
    if (kb < b01)      { src = p0; off = kb; }
    else if (kb < b12) { src = p1; off = kb - b01; }
    else               { src = p2; off = kb - b12; }

    // A tile, CDNA5 16-bit layout: lane group 0 -> K {0..7,16..23},
    // lane group 1 -> K {8..15,24..31}, row m = mbase + (lane&15).
    const bf16* q = src + off + hs * 8;
    const v8bf c0 = *(const v8bf*)(q);         // global_load_b128
    const v8bf c1 = *(const v8bf*)(q + 16);    // global_load_b128
    v16bf a;
    if (g.bn_scale) {
      const int k0 = kb + hs * 8;
#pragma unroll
      for (int i = 0; i < 8; ++i) {
        float f0 = fmaxf((float)c0[i] * Sl[k0 + i]      + Hl[k0 + i],      0.0f);
        float f1 = fmaxf((float)c1[i] * Sl[k0 + 16 + i] + Hl[k0 + 16 + i], 0.0f);
        a[i] = (bf16)f0; a[8 + i] = (bf16)f1;
      }
    } else {
#pragma unroll
      for (int i = 0; i < 8; ++i) { a[i] = c0[i]; a[8 + i] = c1[i]; }
    }

#pragma unroll
    for (int nt = 0; nt < NT; ++nt) {
      const v16bf b = load_b(Wl, g.Kp, nt, kt, lane);
      acc[nt] = __builtin_amdgcn_wmma_f32_16x16x32_bf16(
          /*neg_a=*/false, a, /*neg_b=*/false, b,
          /*c_mod=*/(short)0, acc[nt], /*reuse_a=*/false, /*reuse_b=*/false);
    }
  }

  // Epilogue: bias + store (D layout: lane l vgpr rr -> M = rr + 8*(l>>4), N = l&15)
  const int nlo = lane & 15;
#pragma unroll
  for (int nt = 0; nt < NT; ++nt) {
    const int n   = nt * 16 + nlo;
    const float bia = Bl[n];
#pragma unroll
    for (int rr = 0; rr < 8; ++rr) {
      const int mm = mbase + rr + 8 * hs;
      const float v = acc[nt][rr] + bia;
      if (g.scat_dst) {
        atomicAdd(&g.scat_dst[(size_t)g.scat_idx[mm] * g.ldo + n], v);
      } else {
        const size_t o = (size_t)mm * g.ldo + n;
        if (g.out_bf)  g.out_bf[o]  = (bf16)v;
        if (g.out_f32) g.out_f32[o] = v;
      }
    }
  }
}

// ---------------------------------------------------------------------------
// Small helper kernels
// ---------------------------------------------------------------------------
__global__ void zero_f32(float* p, int n) {
  const int i = blockIdx.x * blockDim.x + threadIdx.x;
  if (i < n) p[i] = 0.0f;
}

// fp32 [M x win] -> bf16 [M x wout], zero-padding columns win..wout-1.
__global__ void pad_convert(const float* __restrict__ in, bf16* __restrict__ out,
                            int M, int win, int wout) {
  const int t = blockIdx.x * blockDim.x + threadIdx.x;
  const int r = t / wout, c = t - r * wout;
  if (r >= M) return;
  out[(size_t)r * wout + c] = (c < win) ? (bf16)in[(size_t)r * win + c] : (bf16)0.0f;
}

// Per-column sum & sum-of-squares over H [M x 128] (training-mode BatchNorm).
__global__ void col_stats(const bf16* __restrict__ H, int M,
                          float* __restrict__ sums, float* __restrict__ sumsq) {
  const int c      = threadIdx.x & 127;
  const int walker = blockIdx.x * (blockDim.x >> 7) + (threadIdx.x >> 7);
  const int nwalk  = gridDim.x * (blockDim.x >> 7);
  float s = 0.0f, s2 = 0.0f;
  for (int r = walker; r < M; r += nwalk) {
    const float v = (float)H[(size_t)r * 128 + c];
    s += v; s2 += v * v;
  }
  atomicAdd(&sums[c], s);
  atomicAdd(&sumsq[c], s2);
}

__global__ void bn_final(const float* sums, const float* sumsq,
                         const float* gam, const float* bet, float invM,
                         float* scale, float* shiftv) {
  const int c = threadIdx.x;  // 128
  const float mu  = sums[c] * invM;
  const float var = sumsq[c] * invM - mu * mu;   // biased variance
  const float sc  = gam[c] * rsqrtf(var + 1e-5f);
  scale[c]  = sc;
  shiftv[c] = bet[c] - mu * sc;
}

// Segment accumulation for scatter_mean: vals [M x 64] bf16 into accum[g][colOff..+63]
__global__ void seg_accum(const bf16* __restrict__ vals, const int* __restrict__ ids,
                          const int* __restrict__ gather, float* __restrict__ accum,
                          int colOff, float* __restrict__ cnt, int M) {
  const int t = blockIdx.x * blockDim.x + threadIdx.x;
  const int r = t >> 6, c = t & 63;
  if (r >= M) return;
  const int b = gather ? ids[gather[r]] : ids[r];
  atomicAdd(&accum[b * 128 + colOff + c], (float)vals[(size_t)r * 64 + c]);
  if (c == 0) atomicAdd(&cnt[b], 1.0f);
}

__global__ void seg_div(const float* __restrict__ accum, const float* __restrict__ cntN,
                        const float* __restrict__ cntE, bf16* __restrict__ out) {
  const int t = blockIdx.x * blockDim.x + threadIdx.x;  // 64*128
  if (t >= GG * 128) return;
  const int g = t >> 7, c = t & 127;
  const float cnt = (c < 64) ? cntN[g] : cntE[g];
  out[t] = (bf16)(accum[t] / fmaxf(cnt, 1.0f));
}

// Final shift head: out[m] = b2 + sum_k relu(H[m,k]) * W2[k]   (Nout=1)
__global__ void shift_final(const bf16* __restrict__ H, const float* __restrict__ W2,
                            const float* __restrict__ b2, float* __restrict__ out, int M) {
  const int m = blockIdx.x * blockDim.x + threadIdx.x;
  if (m >= M) return;
  float s = b2[0];
#pragma unroll 4
  for (int k = 0; k < 128; ++k)
    s += fmaxf((float)H[(size_t)m * 128 + k], 0.0f) * W2[k];
  out[m] = s;
}

// ---------------------------------------------------------------------------
// Host-side helpers
// ---------------------------------------------------------------------------
struct Seg { const bf16* p; const int* g; int w; int ld; };

static void launch_gemm(hipStream_t s, Seg s0, Seg s1, Seg s2, int Kp, int Kw,
                        const float* W, const float* bias,
                        const float* bnsc, const float* bnsh,
                        bf16* obf, float* of32, int ldo,
                        const int* sidx, float* sdst,
                        int Mtiles, int Nout) {
  GemmArgs g;
  g.a0 = s0.p; g.g0 = s0.g; g.w0 = s0.w; g.ld0 = s0.ld;
  g.a1 = s1.p; g.g1 = s1.g; g.w1 = s1.w; g.ld1 = s1.ld;
  g.a2 = s2.p; g.g2 = s2.g; g.w2 = s2.w; g.ld2 = s2.ld;
  g.Kp = Kp; g.W = W; g.Kw = Kw; g.bias = bias;
  g.bn_scale = bnsc; g.bn_shift = bnsh;
  g.out_bf = obf; g.out_f32 = of32; g.ldo = ldo;
  g.scat_idx = sidx; g.scat_dst = sdst;
  g.Mtiles = Mtiles; g.Nout = Nout;
  const int blocks = (Mtiles + 7) / 8;
  const size_t lds = (size_t)Nout * Kp * sizeof(bf16)
                   + (size_t)Nout * sizeof(float)
                   + (size_t)2 * Kp * sizeof(float);
  if (Nout == 128) wmma_gemm<8><<<blocks, 256, lds, s>>>(g);
  else             wmma_gemm<4><<<blocks, 256, lds, s>>>(g);
}

static void zero(hipStream_t s, float* p, int n) {
  zero_f32<<<(n + 255) / 256, 256, 0, s>>>(p, n);
}

// Two-layer MLP with hidden BatchNorm+ReLU (PyG-style).
static void mlp_bn(hipStream_t s, Seg s0, Seg s1, Seg s2, int Kp, int Kw,
                   const float* W0, const float* b0, const float* gam, const float* bet,
                   const float* W1, const float* b1, int Nout2,
                   bf16* hid, int Mrows,
                   bf16* obf, float* of32, const int* sidx, float* sdst,
                   float* stats /* 512 floats: sums,sumsq,scale,shift */) {
  const int Mtiles = Mrows / 16;
  float* sums = stats;       float* sumsq = stats + 128;
  float* sc   = stats + 256; float* sh    = stats + 384;
  zero(s, sums, 256);
  launch_gemm(s, s0, s1, s2, Kp, Kw, W0, b0, nullptr, nullptr,
              hid, nullptr, 128, nullptr, nullptr, Mtiles, 128);
  col_stats<<<512, 256, 0, s>>>(hid, Mrows, sums, sumsq);
  bn_final<<<1, 128, 0, s>>>(sums, sumsq, gam, bet, 1.0f / (float)Mrows, sc, sh);
  Seg hseg{hid, nullptr, 128, 128}, z{nullptr, nullptr, 0, 0};
  launch_gemm(s, hseg, z, z, 128, 128, W1, b1, sc, sh,
              obf, of32, Nout2, sidx, sdst, Mtiles, Nout2);
}

struct MlpP { const float *W0, *b0, *bt0, *g0, *W1, *b1; };
static MlpP mlp_at(void* const* d, int b) {
  // jax tree-flatten: layer dict keys sorted: W, b, bt, g ; final layer: W, b
  return { (const float*)d[b + 0], (const float*)d[b + 1],
           (const float*)d[b + 2], (const float*)d[b + 3],
           (const float*)d[b + 4], (const float*)d[b + 5] };
}

// ---------------------------------------------------------------------------
extern "C" void kernel_launch(void* const* d_in, const int* in_sizes, int n_in,
                              void* d_out, int out_size, void* d_ws, size_t ws_size,
                              hipStream_t stream) {
  (void)in_sizes; (void)n_in; (void)out_size; (void)ws_size;
  // --- parameter leaves (sorted-key jax flatten of the params dict) -------
  // 0..5   edge_enc | 6..29 layers[0]{edge,glob,node1,node2} | 30..53 layers[1]
  // 54..59 node_enc | 60..63 shift (no norm: W0,b0,W1,b1)
  const MlpP edge_enc = mlp_at(d_in, 0);
  const MlpP node_enc = mlp_at(d_in, 54);
  MlpP Ledge[2], Lglob[2], Lnode1[2], Lnode2[2];
  for (int l = 0; l < 2; ++l) {
    const int b = 6 + l * 24;
    Ledge[l]  = mlp_at(d_in, b + 0);
    Lglob[l]  = mlp_at(d_in, b + 6);
    Lnode1[l] = mlp_at(d_in, b + 12);
    Lnode2[l] = mlp_at(d_in, b + 18);
  }
  const float* shift_W0 = (const float*)d_in[60];
  const float* shift_b0 = (const float*)d_in[61];
  const float* shift_W1 = (const float*)d_in[62];
  const float* shift_b1 = (const float*)d_in[63];

  const float* x_in  = (const float*)d_in[64];        // [N,32]
  const int*   eidx  = (const int*)d_in[65];          // [2,E]
  const float* eattr = (const float*)d_in[66];        // [E,16]
  const int*   batch = (const int*)d_in[67];          // [N]
  const int* row = eidx;
  const int* col = eidx + EE;

  // --- d_out layout: shifts [N], x [N,64], ea [E,64], u [G,64] ------------
  float* out_shift = (float*)d_out;
  float* out_x  = out_shift + NN;
  float* out_ea = out_x + (size_t)NN * 64;
  float* out_u  = out_ea + (size_t)EE * 64;

  // --- workspace carve ----------------------------------------------------
  char* w = (char*)d_ws;
  auto carve = [&](size_t bytes) -> char* {
    char* p = w; w += (bytes + 255) & ~(size_t)255; return p;
  };
  bf16* xinbf  = (bf16*)carve((size_t)NN * 32 * 2);
  bf16* eainbf = (bf16*)carve((size_t)EE * 32 * 2);   // padded 16 -> 32 cols
  bf16* xbf    = (bf16*)carve((size_t)NN * 64 * 2);
  bf16* eabf   = (bf16*)carve((size_t)EE * 64 * 2);
  bf16* hbig   = (bf16*)carve((size_t)EE * 128 * 2);  // shared hidden buffer
  float* aggf  = (float*)carve((size_t)NN * 128 * 4);
  bf16* aggbf  = (bf16*)carve((size_t)NN * 128 * 2);
  bf16* ubf    = (bf16*)carve((size_t)GG * 64 * 2);
  float* nmemf = (float*)carve((size_t)GG * 128 * 4);
  bf16* nmembf = (bf16*)carve((size_t)GG * 128 * 2);
  float* cntN  = (float*)carve(GG * 4);
  float* cntE  = (float*)carve(GG * 4);
  float* stats = (float*)carve(512 * 4);

  const Seg z{nullptr, nullptr, 0, 0};

  // --- input conversions to bf16 (pad edge_attr 16 -> 32 columns) ---------
  pad_convert<<<(NN * 32 + 255) / 256, 256, 0, stream>>>(x_in, xinbf, NN, 32, 32);
  pad_convert<<<(EE * 32 + 255) / 256, 256, 0, stream>>>(eattr, eainbf, EE, 16, 32);

  // --- encoders -----------------------------------------------------------
  mlp_bn(stream, Seg{xinbf, nullptr, 32, 32}, z, z, 32, 32,
         node_enc.W0, node_enc.b0, node_enc.g0, node_enc.bt0,
         node_enc.W1, node_enc.b1, 64, hbig, NN,
         xbf, out_x, nullptr, nullptr, stats);
  mlp_bn(stream, Seg{eainbf, nullptr, 32, 32}, z, z, 32, 16,
         edge_enc.W0, edge_enc.b0, edge_enc.g0, edge_enc.bt0,
         edge_enc.W1, edge_enc.b1, 64, hbig, EE,
         eabf, nullptr, nullptr, nullptr, stats);

  // --- message-passing layers ---------------------------------------------
  for (int l = 0; l < 2; ++l) {
    // EdgeModel: ea = MLP(cat(x[row], x[col], ea))
    mlp_bn(stream, Seg{xbf, row, 64, 64}, Seg{xbf, col, 64, 64}, Seg{eabf, nullptr, 64, 64},
           192, 192, Ledge[l].W0, Ledge[l].b0, Ledge[l].g0, Ledge[l].bt0,
           Ledge[l].W1, Ledge[l].b1, 64, hbig, EE,
           eabf, out_ea, nullptr, nullptr, stats);

    // NodeModel part 1: m = MLP(cat(x[row], ea)); agg = segment_sum(m, col)
    zero(stream, aggf, NN * 128);
    mlp_bn(stream, Seg{xbf, row, 64, 64}, Seg{eabf, nullptr, 64, 64}, z,
           128, 128, Lnode1[l].W0, Lnode1[l].b0, Lnode1[l].g0, Lnode1[l].bt0,
           Lnode1[l].W1, Lnode1[l].b1, 128, hbig, EE,
           nullptr, nullptr, col, aggf, stats);        // fused scatter-add epilogue
    pad_convert<<<(NN * 128 + 255) / 256, 256, 0, stream>>>(aggf, aggbf, NN, 128, 128);

    // NodeModel part 2: x = MLP(cat(x, agg))
    mlp_bn(stream, Seg{xbf, nullptr, 64, 64}, Seg{aggbf, nullptr, 128, 128}, z,
           192, 192, Lnode2[l].W0, Lnode2[l].b0, Lnode2[l].g0, Lnode2[l].bt0,
           Lnode2[l].W1, Lnode2[l].b1, 64, hbig, NN,
           xbf, out_x, nullptr, nullptr, stats);

    // GlobalModel: u = MLP(cat(seg_mean(x,batch), seg_mean(ea,batch[row])))
    zero(stream, nmemf, GG * 128);
    zero(stream, cntN, GG);
    zero(stream, cntE, GG);
    seg_accum<<<(NN * 64 + 255) / 256, 256, 0, stream>>>(xbf, batch, nullptr, nmemf, 0, cntN, NN);
    seg_accum<<<(EE * 64 + 255) / 256, 256, 0, stream>>>(eabf, batch, row, nmemf, 64, cntE, EE);
    seg_div<<<(GG * 128 + 255) / 256, 256, 0, stream>>>(nmemf, cntN, cntE, nmembf);
    mlp_bn(stream, Seg{nmembf, nullptr, 128, 128}, z, z, 128, 128,
           Lglob[l].W0, Lglob[l].b0, Lglob[l].g0, Lglob[l].bt0,
           Lglob[l].W1, Lglob[l].b1, 64, hbig, GG,
           ubf, out_u, nullptr, nullptr, stats);
  }

  // --- shift head: MLP([x, u[batch]]) -> [N,1]  (no BatchNorm) -------------
  launch_gemm(stream, Seg{xbf, nullptr, 64, 64}, Seg{ubf, batch, 64, 64}, z,
              128, 128, shift_W0, shift_b0, nullptr, nullptr,
              hbig, nullptr, 128, nullptr, nullptr, NN / 16, 128);
  shift_final<<<(NN + 255) / 256, 256, 0, stream>>>(hbig, shift_W1, shift_b1, out_shift, NN);
}
